// EGNN_67284957659166
// MI455X (gfx1250) — matrix operand order repaired and checked by reading, not compiled, not run on hardware
//
#include <hip/hip_runtime.h>
#include <hip/hip_bf16.h>

#define NN 50000
#define EE 800000
#define LL 5
#define DD 128
#define GG 512
#define TT 10
#define ATOM_ROWS 119
#define NP 50048          // NN padded to 391 * 128 (no guards in GEMM)

typedef __attribute__((ext_vector_type(16))) __bf16 v16bf;
typedef __attribute__((ext_vector_type(4)))  __bf16 v4bf;
typedef __attribute__((ext_vector_type(8)))  float  v8f;
typedef __attribute__((ext_vector_type(4)))  float  v4f;

__device__ __forceinline__ unsigned short f32_to_bf16_bits(float f) {
    unsigned u = __float_as_uint(f);
    u += 0x7FFFu + ((u >> 16) & 1u);   // round-to-nearest-even
    return (unsigned short)(u >> 16);
}
// hardware path: v_cvt_pk_bf16_f32
__device__ __forceinline__ uint2 cvt4_bf16(v4f f) {
    union { v4bf v; uint2 u; } c;
    c.v = __builtin_convertvector(f, v4bf);
    return c.u;
}

// ---------------- atom encoder: h[n,d] = sum_t emb[t, x[n,t], d] ----------------
__global__ __launch_bounds__(128)
void atom_encoder_kernel(const int* __restrict__ x, const float* __restrict__ emb,
                         float* __restrict__ h) {
    __shared__ int xi[9];
    int n = blockIdx.x;
    if (threadIdx.x < 9) xi[threadIdx.x] = x[n * 9 + threadIdx.x];
    __syncthreads();
    int d = threadIdx.x;
    float acc = 0.f;
#pragma unroll
    for (int t = 0; t < 9; ++t)
        acc += emb[((size_t)(t * ATOM_ROWS + xi[t])) * DD + d];
    h[(size_t)n * DD + d] = acc;
}

// ---------------- weights: f32 [K,Ncols] -> bf16 transposed [Ncols,K] ----------------
__global__ void convert_transpose_kernel(const float* __restrict__ W, int K, int Ncols,
                                         unsigned short* __restrict__ Wt) {
    int i = blockIdx.x * 256 + threadIdx.x;
    if (i >= K * Ncols) return;
    int k = i / Ncols, n = i - k * Ncols;
    Wt[(size_t)n * K + k] = f32_to_bf16_bits(W[i]);
}

// ---------------- z = (1+eps[layer]) * h  (float4) ----------------
__global__ void init_z_kernel(const float4* __restrict__ h, const float* __restrict__ eps,
                              int layer, float4* __restrict__ z, int n4) {
    int i = blockIdx.x * blockDim.x + threadIdx.x;
    if (i >= n4) return;
    float s = 1.f + eps[layer];
    float4 v = h[i];
    v.x *= s; v.y *= s; v.z *= s; v.w *= s;
    z[i] = v;
}

// ---------------- z[dst] += relu(h[src]) : 32 lanes/edge, float4/lane ----------------
__global__ __launch_bounds__(256)
void scatter_kernel(const float* __restrict__ h, const int* __restrict__ src,
                    const int* __restrict__ dst, float* __restrict__ z) {
    int t = blockIdx.x * 256 + threadIdx.x;
    int e = t >> 5, lane = t & 31;
    if (e >= EE) return;
    int s = src[e], d = dst[e];
    const float4 v = *(const float4*)(h + (size_t)s * DD + lane * 4);
    float* zp = z + (size_t)d * DD + lane * 4;
    atomicAdd(zp + 0, v.x > 0.f ? v.x : 0.f);
    atomicAdd(zp + 1, v.y > 0.f ? v.y : 0.f);
    atomicAdd(zp + 2, v.z > 0.f ? v.z : 0.f);
    atomicAdd(zp + 3, v.w > 0.f ? v.w : 0.f);
}

// ---------------- bf16 WMMA GEMM (LDS-free): Out = act(A) @ W + bias ----------------
// A:  [Mp,K] f32, Mp multiple of 128 (no guards).
// Wt: [Ncols,K] bf16 (pre-transposed) -> B fragments are contiguous b128 loads.
// TRANSFORM=1 fuses per-column affine+relu into the A fragment load.
// block = 8 waves (wave32); tile 128 rows x 64 cols; K-step 32.
template <int TRANSFORM>
__global__ __launch_bounds__(256)
void gemm_bn_kernel(const float* __restrict__ A, int K, int Ncols,
                    const unsigned short* __restrict__ Wt, const float* __restrict__ bias,
                    const float* __restrict__ a_scale, const float* __restrict__ a_shift,
                    float* __restrict__ Out) {
    const int tid  = threadIdx.x;
    const int wave = tid >> 5;          // 8 waves / block
    const int lane = tid & 31;
    const int half = lane >> 4;         // CDNA5 16-bit layouts: lane half h holds k 8h..8h+7 of each 16-k half
    const int r    = lane & 15;
    const int row  = blockIdx.x * 128 + wave * 16 + r;   // this lane's A row
    const int colBase = blockIdx.y * 64;

    const float* __restrict__ arow = A + (size_t)row * K;

    v8f acc0 = {}, acc1 = {}, acc2 = {}, acc3 = {};

    for (int k0 = 0; k0 < K; k0 += 32) {
        // ---- A fragment: two contiguous 8-float K-ranges of this lane's row ----
        const int kA0 = k0 + half * 8;         // covers u[0..3]
        const int kA1 = k0 + 16 + half * 8;    // covers u[4..7]
        v4f a0 = *(const v4f*)(arow + kA0);
        v4f a1 = *(const v4f*)(arow + kA0 + 4);
        v4f a2 = *(const v4f*)(arow + kA1);
        v4f a3 = *(const v4f*)(arow + kA1 + 4);
        if (TRANSFORM) {
            const v4f z4 = {0.f, 0.f, 0.f, 0.f};
            a0 = __builtin_elementwise_max(a0 * *(const v4f*)(a_scale + kA0)     + *(const v4f*)(a_shift + kA0),     z4);
            a1 = __builtin_elementwise_max(a1 * *(const v4f*)(a_scale + kA0 + 4) + *(const v4f*)(a_shift + kA0 + 4), z4);
            a2 = __builtin_elementwise_max(a2 * *(const v4f*)(a_scale + kA1)     + *(const v4f*)(a_shift + kA1),     z4);
            a3 = __builtin_elementwise_max(a3 * *(const v4f*)(a_scale + kA1 + 4) + *(const v4f*)(a_shift + kA1 + 4), z4);
        }
        union { v16bf v; uint2 u2[4]; } af;
        af.u2[0] = cvt4_bf16(a0);
        af.u2[1] = cvt4_bf16(a1);
        af.u2[2] = cvt4_bf16(a2);
        af.u2[3] = cvt4_bf16(a3);

        // ---- B fragments: two contiguous 8-value bf16 K-ranges of this lane's column ----
#define LOAD_B(frag, nt)                                                            \
        {                                                                           \
            const unsigned short* bcol =                                            \
                Wt + (size_t)(colBase + (nt)*16 + r) * K + k0 + half * 8;           \
            frag.u4[0] = *(const uint4*)bcol;                                       \
            frag.u4[1] = *(const uint4*)(bcol + 16);                                \
        }
        union { v16bf v; uint4 u4[2]; } bf0, bf1, bf2, bf3;
        LOAD_B(bf0, 0) LOAD_B(bf1, 1) LOAD_B(bf2, 2) LOAD_B(bf3, 3)
#undef LOAD_B

        acc0 = __builtin_amdgcn_wmma_f32_16x16x32_bf16(false, af.v, false, bf0.v, (short)0, acc0, false, false);
        acc1 = __builtin_amdgcn_wmma_f32_16x16x32_bf16(false, af.v, false, bf1.v, (short)0, acc1, false, false);
        acc2 = __builtin_amdgcn_wmma_f32_16x16x32_bf16(false, af.v, false, bf2.v, (short)0, acc2, false, false);
        acc3 = __builtin_amdgcn_wmma_f32_16x16x32_bf16(false, af.v, false, bf3.v, (short)0, acc3, false, false);
    }

    // store D (+bias). D layout: VGPR j -> M = j + 8*half, N = r. Unconditional (padded rows).
    const int rowD = blockIdx.x * 128 + wave * 16 + half * 8;
#pragma unroll
    for (int nt = 0; nt < 4; ++nt) {
        v8f acc = (nt == 0) ? acc0 : (nt == 1) ? acc1 : (nt == 2) ? acc2 : acc3;
        int col = colBase + nt * 16 + r;
        float b = bias[col];
#pragma unroll
        for (int j = 0; j < 8; ++j)
            Out[(size_t)(rowD + j) * Ncols + col] = acc[j] + b;
    }
}

// ---------------- per-column sum / sumsq (blockDim.x == C) ----------------
__global__ void col_stats_kernel(const float* __restrict__ Z, int M, int C,
                                 int rowsPerBlock, float* __restrict__ stats) {
    int c = threadIdx.x;
    int r0 = blockIdx.x * rowsPerBlock;
    int r1 = r0 + rowsPerBlock; if (r1 > M) r1 = M;
    float s = 0.f, q = 0.f;
    for (int r = r0; r < r1; ++r) {
        float v = Z[(size_t)r * C + c];
        s += v; q += v * v;
    }
    atomicAdd(&stats[c], s);
    atomicAdd(&stats[C + c], q);
}

// ---------------- BN finalize: scale/shift per column ----------------
__global__ void bn_finalize_kernel(const float* __restrict__ stats, const float* __restrict__ g,
                                   const float* __restrict__ be, int C, float invM,
                                   float* __restrict__ ab) {
    int c = blockIdx.x * blockDim.x + threadIdx.x;
    if (c >= C) return;
    float mean = stats[c] * invM;
    float var  = stats[C + c] * invM - mean * mean;
    float a = g[c] * rsqrtf(var + 1e-5f);
    ab[c] = a;
    ab[C + c] = be[c] - mean * a;
}

// ---------------- h = relu(a*x + b) elementwise (float4), C == 128 ----------------
__global__ void bn_relu_kernel(const float4* __restrict__ X, const float* __restrict__ ab,
                               float4* __restrict__ Y, int n4) {
    int i = blockIdx.x * blockDim.x + threadIdx.x;
    if (i >= n4) return;
    int c = (i & 31) * 4;                              // (i*4) mod 128
    const float4 a = *(const float4*)(ab + c);
    const float4 b = *(const float4*)(ab + DD + c);
    float4 v = X[i];
    v.x = fmaxf(a.x * v.x + b.x, 0.f);
    v.y = fmaxf(a.y * v.y + b.y, 0.f);
    v.z = fmaxf(a.z * v.z + b.z, 0.f);
    v.w = fmaxf(a.w * v.w + b.w, 0.f);
    Y[i] = v;
}

// ---------------- pooling accumulate ----------------
__global__ __launch_bounds__(256)
void pool_acc_kernel(const float* __restrict__ h, const int* __restrict__ batch,
                     float* __restrict__ sums, float* __restrict__ cnt) {
    int t = blockIdx.x * 256 + threadIdx.x;
    int n = t >> 5, lane = t & 31;
    if (n >= NN) return;
    int g = batch[n];
    const float4 v = *(const float4*)(h + (size_t)n * DD + lane * 4);
    float* p = sums + (size_t)g * DD + lane * 4;
    atomicAdd(p + 0, v.x); atomicAdd(p + 1, v.y);
    atomicAdd(p + 2, v.z); atomicAdd(p + 3, v.w);
    if (lane == 0) atomicAdd(&cnt[g], 1.0f);
}

// ---------------- out[g,:] = (sums[g]/cnt[g]) @ Wout + bout ----------------
__global__ __launch_bounds__(128)
void pool_out_kernel(const float* __restrict__ sums, const float* __restrict__ cnt,
                     const float* __restrict__ Wout, const float* __restrict__ bout,
                     float* __restrict__ out) {
    __shared__ float red[128];
    int g = blockIdx.x, d = threadIdx.x;
    float c = cnt[g]; if (c < 1.f) c = 1.f;
    float v = sums[(size_t)g * DD + d] / c;
    for (int t = 0; t < TT; ++t) {
        red[d] = v * Wout[d * TT + t];
        __syncthreads();
        for (int s = 64; s > 0; s >>= 1) {
            if (d < s) red[d] += red[d + s];
            __syncthreads();
        }
        if (d == 0) out[g * TT + t] = red[0] + bout[t];
        __syncthreads();
    }
}

extern "C" void kernel_launch(void* const* d_in, const int* in_sizes, int n_in,
                              void* d_out, int out_size, void* d_ws, size_t ws_size,
                              hipStream_t stream) {
    const int*   x        = (const int*)  d_in[0];
    const int*   eidx     = (const int*)  d_in[1];   // [2,E]
    const int*   batch    = (const int*)  d_in[2];
    const float* atom_emb = (const float*)d_in[3];
    const float* W1       = (const float*)d_in[4];   // [L,128,256]
    const float* b1       = (const float*)d_in[5];   // [L,256]
    const float* g1       = (const float*)d_in[6];
    const float* be1      = (const float*)d_in[7];
    const float* W2       = (const float*)d_in[8];   // [L,256,128]
    const float* b2       = (const float*)d_in[9];   // [L,128]
    const float* eps      = (const float*)d_in[10];  // [L]
    const float* bn_g     = (const float*)d_in[11];
    const float* bn_b     = (const float*)d_in[12];
    const float* Wout     = (const float*)d_in[13];  // [128,10]
    const float* bout     = (const float*)d_in[14];
    float* out = (float*)d_out;

    const int* src = eidx;
    const int* dst = eidx + EE;

    // workspace carve-up; node buffers padded to NP rows so GEMM needs no guards
    float* h      = (float*)d_ws;              // NP*128
    float* z      = h  + (size_t)NP * DD;      // NP*128
    float* z1     = z  + (size_t)NP * DD;      // NP*256
    float* h2     = z1 + (size_t)NP * 2 * DD;  // NP*128
    float* stats1 = h2 + (size_t)NP * DD;      // 512
    float* ab1    = stats1 + 512;              // 512
    float* stats2 = ab1 + 512;                 // 256
    float* ab2    = stats2 + 256;              // 256
    float* sums   = ab2 + 256;                 // 512*128
    float* cnt    = sums + (size_t)GG * DD;    // 512
    unsigned short* w1t = (unsigned short*)(cnt + GG);          // L * 256*128 bf16
    unsigned short* w2t = w1t + (size_t)LL * 2 * DD * DD;       // L * 128*256 bf16

    const int n4 = NN * DD / 4;                // 1.6M float4
    const int rowBlocks = NP / 128;            // 391
    const int statRows = 256;
    const int statBlocks = (NN + statRows - 1) / statRows;
    const int wElems = 2 * DD * DD;            // 32768 per layer per matrix

    // one-time weight convert+transpose (bf16, column-major -> contiguous B fragments)
    for (int i = 0; i < LL; ++i) {
        convert_transpose_kernel<<<(wElems + 255) / 256, 256, 0, stream>>>(
            W1 + (size_t)i * wElems, DD, 2 * DD, w1t + (size_t)i * wElems);
        convert_transpose_kernel<<<(wElems + 255) / 256, 256, 0, stream>>>(
            W2 + (size_t)i * wElems, 2 * DD, DD, w2t + (size_t)i * wElems);
    }

    // atom encoder
    atom_encoder_kernel<<<NN, 128, 0, stream>>>(x, atom_emb, h);

    for (int i = 0; i < LL; ++i) {
        // z = (1+eps)h ; z += scatter(relu(h[src]))
        init_z_kernel<<<(n4 + 255) / 256, 256, 0, stream>>>((const float4*)h, eps, i,
                                                            (float4*)z, n4);
        scatter_kernel<<<(EE * 32 + 255) / 256, 256, 0, stream>>>(h, src, dst, z);

        // z1 = z @ W1[i] + b1[i]      [N,256]
        gemm_bn_kernel<0><<<dim3(rowBlocks, 4), 256, 0, stream>>>(
            z, DD, 2 * DD, w1t + (size_t)i * wElems, b1 + (size_t)i * 2 * DD,
            nullptr, nullptr, z1);

        // BN1 stats + finalize
        hipMemsetAsync(stats1, 0, 512 * sizeof(float), stream);
        col_stats_kernel<<<statBlocks, 2 * DD, 0, stream>>>(z1, NN, 2 * DD, statRows, stats1);
        bn_finalize_kernel<<<1, 2 * DD, 0, stream>>>(stats1, g1 + (size_t)i * 2 * DD,
                                                     be1 + (size_t)i * 2 * DD, 2 * DD,
                                                     1.0f / NN, ab1);

        // h2 = relu(bn1(z1)) @ W2[i] + b2[i]   [N,128]  (affine+relu fused into A fragments)
        gemm_bn_kernel<1><<<dim3(rowBlocks, 2), 256, 0, stream>>>(
            z1, 2 * DD, DD, w2t + (size_t)i * wElems, b2 + (size_t)i * DD,
            ab1, ab1 + 2 * DD, h2);

        if (i < LL - 1) {
            // outer BN + relu -> h
            hipMemsetAsync(stats2, 0, 256 * sizeof(float), stream);
            col_stats_kernel<<<statBlocks, DD, 0, stream>>>(h2, NN, DD, statRows, stats2);
            bn_finalize_kernel<<<1, DD, 0, stream>>>(stats2, bn_g + (size_t)i * DD,
                                                     bn_b + (size_t)i * DD, DD,
                                                     1.0f / NN, ab2);
            bn_relu_kernel<<<(n4 + 255) / 256, 256, 0, stream>>>((const float4*)h2, ab2,
                                                                 (float4*)h, n4);
        }
    }

    // mean pool + output GEMM (reads h2 from layer L-1)
    hipMemsetAsync(sums, 0, (size_t)GG * DD * sizeof(float), stream);
    hipMemsetAsync(cnt, 0, GG * sizeof(float), stream);
    pool_acc_kernel<<<(NN * 32 + 255) / 256, 256, 0, stream>>>(h2, batch, sums, cnt);
    pool_out_kernel<<<GG, 128, 0, stream>>>(sums, cnt, Wout, bout, out);
}
